// GNN_AttentiveFP_18279380811837
// MI455X (gfx1250) — matrix operand order repaired and checked by reading, not compiled, not run on hardware
//
#include <hip/hip_runtime.h>
#include <math.h>

#define NN 50000
#define EE 800000
#define GG 1024
#define HH 64

typedef __attribute__((ext_vector_type(16))) _Float16 v16h;
typedef __attribute__((ext_vector_type(8)))  float    v8f;

// ---------------- WMMA helpers (CDNA5 16x16x32 f16 -> f32) ----------------

__device__ __forceinline__ v8f wmma_f16(v16h a, v16h b, v8f c) {
  return __builtin_amdgcn_wmma_f32_16x16x32_f16(
      false, a, false, b, (short)0, c, false, false);
}

// A fragment, 16x32 f16, lane l: row m=l&15, half=(l>>4): K = h*16..h*16+15
__device__ __forceinline__ v16h load_a_frag(const float* __restrict__ X, int ldx,
                                            int row, int k0) {
  v16h a;
  const float* p = X + (long long)row * ldx + k0;
#pragma unroll
  for (int i = 0; i < 8; i++) a[i] = (_Float16)p[i];
#pragma unroll
  for (int i = 0; i < 8; i++) a[8 + i] = (_Float16)p[8 + i];
  return a;
}

// B fragment, 32x16 f16 (B[k,n] = W[n,k]), lane l: col n, half=(l>>4): K = h*16..h*16+15
__device__ __forceinline__ v16h load_b_frag(const float* __restrict__ W, int ldw,
                                            int n, int kb) {
  v16h b;
  const float* p = W + (long long)n * ldw + kb;
#pragma unroll
  for (int i = 0; i < 16; i++) b[i] = (_Float16)p[i];
  return b;
}

__device__ __forceinline__ float leaky01(float v) { return v > 0.f ? v : 0.01f * v; }

// ---------------- Generic WMMA GEMM:  Y = act(X @ W^T + b) ----------------
// Precondition: rows % 16 == 0 (true for N=50000, G=1024) -> no per-element guards.
// B fragments hoisted in registers; each wave processes RTILES row-tiles of 16.
// ACT: 0=none, 1=leaky_relu(0.01), 2=relu
template <int K, int OUT, int ACT, bool BIAS, int RTILES>
__global__ void __launch_bounds__(256)
gemm_act_kernel(const float* __restrict__ X,
                const float* __restrict__ W, int ldw,
                const float* __restrict__ bias,
                float* __restrict__ Y, int rows) {
  constexpr int KT = K / 32, NT = OUT / 16;
  const int wave = threadIdx.x >> 5;
  const int lane = threadIdx.x & 31;
  const int m = lane & 15, half = lane >> 4;

  // hoist all weight fragments into registers (converted once)
  v16h bfrag[KT][NT];
#pragma unroll
  for (int kt = 0; kt < KT; kt++)
#pragma unroll
    for (int nt = 0; nt < NT; nt++)
      bfrag[kt][nt] = load_b_frag(W, ldw, nt * 16 + m, kt * 32 + half * 16);

  float bv[NT];
#pragma unroll
  for (int nt = 0; nt < NT; nt++) bv[nt] = BIAS ? bias[nt * 16 + m] : 0.f;

  const int tile0 = (blockIdx.x * (blockDim.x >> 5) + wave) * RTILES;
#pragma unroll 1
  for (int tr = 0; tr < RTILES; tr++) {
    const int rowbase = (tile0 + tr) * 16;
    if (rowbase >= rows) break;  // wave-uniform: EXEC stays full for WMMA

    v8f acc[NT];
#pragma unroll
    for (int t = 0; t < NT; t++) acc[t] = (v8f)(0.0f);

#pragma unroll
    for (int kt = 0; kt < KT; kt++) {
      v16h a = load_a_frag(X, K, rowbase + m, kt * 32 + half * 16);
#pragma unroll
      for (int nt = 0; nt < NT; nt++) acc[nt] = wmma_f16(a, bfrag[kt][nt], acc[nt]);
    }

#pragma unroll
    for (int nt = 0; nt < NT; nt++) {
      const int n = nt * 16 + m;
#pragma unroll
      for (int r = 0; r < 8; r++) {
        const int row = rowbase + r + half * 8;
        float v = acc[nt][r] + bv[nt];
        if (ACT == 1) v = leaky01(v);
        if (ACT == 2) v = v > 0.f ? v : 0.f;
        Y[(long long)row * OUT + n] = v;
      }
    }
  }
}

// ---------------- Fused GRU cell (WMMA, LDS-staged f16 weights) ----------------
// Precondition: rows % 16 == 0. gi = Hin@Wih^T (+bih), gh = Hx@Whh^T (+bhh);
// gates r,z,n all in-register; Wih/Whh pre-converted to f16 in LDS in B layout.
template <bool RELU>
__global__ void __launch_bounds__(256)
gru_fused_kernel(const float* __restrict__ Hin,
                 const float* __restrict__ Hx,
                 const float* __restrict__ Wih,
                 const float* __restrict__ Whh,
                 const float* __restrict__ bih,
                 const float* __restrict__ bhh,
                 float* __restrict__ Y, int rows) {
  constexpr int NT = 12;  // 192 outputs
  // sB[w][kt][nt][lane][16] : 2*2*12*32*16 halves = 48 KB
  __shared__ __align__(32) _Float16 sB[2 * 2 * NT * 32 * 16];
  __shared__ float sBi[192], sBh[192];

  for (int it = threadIdx.x; it < 2 * 2 * NT * 32; it += blockDim.x) {
    const int l = it & 31;
    int rest = it >> 5;
    const int nt = rest % NT; rest /= NT;
    const int kt = rest & 1;
    const int w  = rest >> 1;
    const float* Wp = (w ? Whh : Wih) +
                      (long long)(nt * 16 + (l & 15)) * HH + kt * 32 + (l >> 4) * 16;
    _Float16* dp = &sB[(((w * 2 + kt) * NT + nt) * 32 + l) * 16];
#pragma unroll
    for (int j = 0; j < 16; j++) dp[j] = (_Float16)Wp[j];
  }
  for (int i = threadIdx.x; i < 192; i += blockDim.x) {
    sBi[i] = bih[i];
    sBh[i] = bhh[i];
  }
  __syncthreads();

  const int wave = threadIdx.x >> 5;
  const int lane = threadIdx.x & 31;
  const int rowbase = (blockIdx.x * (blockDim.x >> 5) + wave) * 16;
  if (rowbase >= rows) return;  // wave-uniform
  const int m = lane & 15, half = lane >> 4;

  v8f gi[NT], gh[NT];
#pragma unroll
  for (int t = 0; t < NT; t++) { gi[t] = (v8f)(0.0f); gh[t] = (v8f)(0.0f); }

#pragma unroll
  for (int kt = 0; kt < 2; kt++) {
    v16h a1 = load_a_frag(Hin, HH, rowbase + m, kt * 32 + half * 16);
    v16h a2 = load_a_frag(Hx,  HH, rowbase + m, kt * 32 + half * 16);
#pragma unroll
    for (int nt = 0; nt < NT; nt++) {
      v16h b1 = *(const v16h*)&sB[(((0 * 2 + kt) * NT + nt) * 32 + lane) * 16];
      gi[nt] = wmma_f16(a1, b1, gi[nt]);
      v16h b2 = *(const v16h*)&sB[(((1 * 2 + kt) * NT + nt) * 32 + lane) * 16];
      gh[nt] = wmma_f16(a2, b2, gh[nt]);
    }
  }

#pragma unroll
  for (int nt = 0; nt < 4; nt++) {
    const int f = nt * 16 + m;
#pragma unroll
    for (int r = 0; r < 8; r++) {
      const int row = rowbase + r + half * 8;
      float sr = gi[nt][r] + sBi[f] + gh[nt][r] + sBh[f];
      float sz = gi[nt + 4][r] + sBi[64 + f] + gh[nt + 4][r] + sBh[64 + f];
      float rg = 1.f / (1.f + __expf(-sr));
      float zg = 1.f / (1.f + __expf(-sz));
      float sn = gi[nt + 8][r] + sBi[128 + f] + rg * (gh[nt + 8][r] + sBh[128 + f]);
      float ng = tanhf(sn);
      float hx = Hx[(long long)row * HH + f];
      float v = (1.f - zg) * ng + zg * hx;
      if (RELU) v = v > 0.f ? v : 0.f;
      Y[(long long)row * HH + f] = v;
    }
  }
}

// ---------------- Elementwise / segment kernels ----------------

__device__ __forceinline__ int f2oi(float f) {
  int i = __float_as_int(f);
  return i >= 0 ? i : (i ^ 0x7fffffff);
}
__device__ __forceinline__ float oi2f(int m_) {
  int i = m_ >= 0 ? m_ : (m_ ^ 0x7fffffff);
  return __int_as_float(i);
}

__global__ void fill_f32_kernel(float* p, float v, int n) {
  int i = blockIdx.x * blockDim.x + threadIdx.x;
  if (i < n) p[i] = v;
}
__global__ void fill_i32_kernel(int* p, int v, int n) {
  int i = blockIdx.x * blockDim.x + threadIdx.x;
  if (i < n) p[i] = v;
}

// node dot products against one or two attention vectors
__global__ void node_dot_kernel(const float* __restrict__ F,
                                const float* __restrict__ att1,
                                const float* __restrict__ att2,
                                float* __restrict__ o1, float* __restrict__ o2,
                                int rows) {
  __shared__ float a1[HH], a2[HH];
  if (threadIdx.x < HH) {
    a1[threadIdx.x] = att1[threadIdx.x];
    a2[threadIdx.x] = att2 ? att2[threadIdx.x] : 0.f;
  }
  __syncthreads();
  int n = blockIdx.x * blockDim.x + threadIdx.x;
  if (n >= rows) return;
  const float* f = F + (long long)n * HH;
  float s1 = 0.f, s2 = 0.f;
#pragma unroll
  for (int k = 0; k < HH; k++) {
    float v = f[k];
    s1 += v * a1[k];
    s2 += v * a2[k];
  }
  o1[n] = s1;
  if (o2) o2[n] = s2;
}

// GATEConv edge logits: a_e = leaky( sum_n leaky(s1[src][n] + ea.W1b[n]) * att_l[n] + gr[dst] )
__global__ void gate_edge_logits_kernel(const float* __restrict__ s1,
                                        const float* __restrict__ gr,
                                        const float* __restrict__ edge_attr,
                                        const int* __restrict__ src,
                                        const int* __restrict__ dst,
                                        const float* __restrict__ W1,  // [64,80]
                                        const float* __restrict__ att_l,
                                        float* __restrict__ aE, int cnt) {
  __shared__ float sW[HH * 16];
  __shared__ float sA[HH];
  for (int i = threadIdx.x; i < HH * 16; i += blockDim.x) {
    int n = i >> 4, k = i & 15;
    sW[i] = W1[n * 80 + 64 + k];
  }
  for (int i = threadIdx.x; i < HH; i += blockDim.x) sA[i] = att_l[i];
  __syncthreads();
  int e = blockIdx.x * blockDim.x + threadIdx.x;
  if (e >= cnt) return;
  const int s = src[e], d = dst[e];
  const float* sp = s1 + (long long)s * HH;
  __builtin_prefetch(sp, 0, 0);  // gathered row, L2-resident -> global_prefetch_b8
  float ea[16];
#pragma unroll
  for (int k = 0; k < 16; k++) ea[k] = edge_attr[(long long)e * 16 + k];
  float a = 0.f;
#pragma unroll 4
  for (int n = 0; n < HH; n++) {
    float h = sp[n];
#pragma unroll
    for (int k = 0; k < 16; k++) h += ea[k] * sW[n * 16 + k];
    a += leaky01(h) * sA[n];
  }
  aE[e] = leaky01(a + gr[d]);
}

// GAT edge logits: a_e = leaky(asn[src] + adn[dst])
__global__ void edge_ss_logits_kernel(const float* __restrict__ asn,
                                      const float* __restrict__ adn,
                                      const int* __restrict__ src,
                                      const int* __restrict__ dst,
                                      float* __restrict__ aE, int cnt) {
  int e = blockIdx.x * blockDim.x + threadIdx.x;
  if (e >= cnt) return;
  aE[e] = leaky01(asn[src[e]] + adn[dst[e]]);
}

// mol logits per node: aN[n] = leaky(asn[n] + adg[batch[n]])
__global__ void mol_node_logits_kernel(const float* __restrict__ asn,
                                       const float* __restrict__ adg,
                                       const int* __restrict__ batch,
                                       float* __restrict__ aN, int cnt) {
  int n = blockIdx.x * blockDim.x + threadIdx.x;
  if (n >= cnt) return;
  aN[n] = leaky01(asn[n] + adg[batch[n]]);
}

__global__ void seg_max_kernel(const float* __restrict__ v,
                               const int* __restrict__ seg,
                               int* __restrict__ amaxI, int cnt) {
  int i = blockIdx.x * blockDim.x + threadIdx.x;
  if (i >= cnt) return;
  atomicMax(&amaxI[seg[i]], f2oi(v[i]));
}

__global__ void seg_expsum_kernel(const float* __restrict__ v,
                                  const int* __restrict__ seg,
                                  const int* __restrict__ amaxI,
                                  float* __restrict__ ssum,
                                  float* __restrict__ eExp, int cnt) {
  int i = blockIdx.x * blockDim.x + threadIdx.x;
  if (i >= cnt) return;
  int s = seg[i];
  float mf = oi2f(amaxI[s]);
  if (!(mf > -1e38f && mf < 1e38f)) mf = 0.f;  // mimic reference isfinite guard
  float e = __expf(v[i] - mf);
  eExp[i] = e;
  atomicAdd(&ssum[s], e);
}

// scatter: agg[dst[e], f] += feat[srcrow, f] * eExp[e] / (ssum[dst]+1e-16)
// src==nullptr -> srcrow = e (identity, used for mol readout); 2 features per thread
__global__ void gat_scatter_kernel(const float* __restrict__ feat,
                                   const float* __restrict__ eExp,
                                   const float* __restrict__ ssum,
                                   const int* __restrict__ src,
                                   const int* __restrict__ dst,
                                   float* __restrict__ agg, int cnt) {
  long long idx = (long long)blockIdx.x * blockDim.x + threadIdx.x;
  if (idx >= (long long)cnt * 32) return;
  int e = (int)(idx >> 5), f = (int)(idx & 31) * 2;
  int d = dst[e];
  int s = src ? src[e] : e;
  float alpha = eExp[e] / (ssum[d] + 1e-16f);
  const float* fs = feat + (long long)s * HH + f;
  float* ag = agg + (long long)d * HH + f;
  atomicAdd(&ag[0], fs[0] * alpha);
  atomicAdd(&ag[1], fs[1] * alpha);
}

__global__ void elu_bias_kernel(float* __restrict__ buf,
                                const float* __restrict__ b, int rows) {
  long long idx = (long long)blockIdx.x * blockDim.x + threadIdx.x;
  if (idx >= (long long)rows * HH) return;
  float v = buf[idx] + b[idx & 63];
  buf[idx] = v > 0.f ? v : (__expf(v) - 1.f);
}

__global__ void pool_kernel(const float* __restrict__ x,
                            const int* __restrict__ batch,
                            float* __restrict__ out, int rows) {
  long long idx = (long long)blockIdx.x * blockDim.x + threadIdx.x;
  if (idx >= (long long)rows * 32) return;
  int n = (int)(idx >> 5), f = (int)(idx & 31) * 2;
  int g = batch[n];
  atomicAdd(&out[(long long)g * HH + f],     x[(long long)n * HH + f]);
  atomicAdd(&out[(long long)g * HH + f + 1], x[(long long)n * HH + f + 1]);
}

__global__ void relu_kernel(float* __restrict__ p, int n) {
  int i = blockIdx.x * blockDim.x + threadIdx.x;
  if (i < n) { float v = p[i]; p[i] = v > 0.f ? v : 0.f; }
}

__global__ void lin2_kernel(const float* __restrict__ out,
                            const float* __restrict__ W,
                            const float* __restrict__ b,
                            float* __restrict__ y, int rows) {
  int g = blockIdx.x * blockDim.x + threadIdx.x;
  if (g >= rows) return;
  const float* o = out + (long long)g * HH;
  float s = 0.f;
#pragma unroll
  for (int k = 0; k < HH; k++) s += o[k] * W[k];
  y[g] = s + b[0];
}

// ---------------- Host-side orchestration ----------------

static inline unsigned cdiv(long long a, long long b) { return (unsigned)((a + b - 1) / b); }

extern "C" void kernel_launch(void* const* d_in, const int* in_sizes, int n_in,
                              void* d_out, int out_size, void* d_ws, size_t ws_size,
                              hipStream_t stream) {
  (void)in_sizes; (void)n_in; (void)out_size; (void)ws_size;
  const float* x         = (const float*)d_in[0];
  const int*   ei        = (const int*)d_in[1];
  const float* edge_attr = (const float*)d_in[2];
  const int*   batch     = (const int*)d_in[3];
  const float* lin1_W = (const float*)d_in[4];
  const float* lin1_b = (const float*)d_in[5];
  const float* gate_att_l = (const float*)d_in[6];
  const float* gate_att_r = (const float*)d_in[7];
  const float* gate_W1 = (const float*)d_in[8];
  const float* gate_W2 = (const float*)d_in[9];
  const float* gate_b  = (const float*)d_in[10];
  const float* gru0_Wih = (const float*)d_in[11];
  const float* gru0_Whh = (const float*)d_in[12];
  const float* gru0_bih = (const float*)d_in[13];
  const float* gru0_bhh = (const float*)d_in[14];
  const float* gru1_Wih = (const float*)d_in[15];
  const float* gru1_Whh = (const float*)d_in[16];
  const float* gru1_bih = (const float*)d_in[17];
  const float* gru1_bhh = (const float*)d_in[18];
  const float* molg_Wih = (const float*)d_in[19];
  const float* molg_Whh = (const float*)d_in[20];
  const float* molg_bih = (const float*)d_in[21];
  const float* molg_bhh = (const float*)d_in[22];
  const float* atom_W = (const float*)d_in[23];
  const float* atom_att_src = (const float*)d_in[24];
  const float* atom_att_dst = (const float*)d_in[25];
  const float* atom_b = (const float*)d_in[26];
  const float* mol_W = (const float*)d_in[27];
  const float* mol_att_src = (const float*)d_in[28];
  const float* mol_att_dst = (const float*)d_in[29];
  const float* mol_b = (const float*)d_in[30];
  const float* lin2_W = (const float*)d_in[31];
  const float* lin2_b = (const float*)d_in[32];

  const int* src = ei;
  const int* dst = ei + EE;

  // workspace layout (floats)
  float* ws = (float*)d_ws;
  size_t o = 0;
  float* NB0 = ws + o; o += (size_t)NN * HH;   // x0 -> x2
  float* NB1 = ws + o; o += (size_t)NN * HH;   // s1 -> x1
  float* NB2 = ws + o; o += (size_t)NN * HH;   // xW2 -> xs(atom) -> xs(mol)
  float* NB3 = ws + o; o += (size_t)NN * HH;   // agg / h
  float* V1  = ws + o; o += NN;                // gr / asn
  float* V2  = ws + o; o += NN;                // adn / aN(mol)
  float* V3  = ws + o; o += NN;                // eExp (mol, per node)
  float* SUM = ws + o; o += NN;                // segment sums (node-level)
  int*   VI  = (int*)(ws + o); o += NN;        // ordered-int segment max
  float* E1  = ws + o; o += EE;                // edge logits
  float* E2  = ws + o; o += EE;                // edge exp
  float* OUT = ws + o; o += (size_t)GG * HH;   // molecule state
  float* GD  = ws + o; o += (size_t)GG * HH;   // xd
  float* GA  = ws + o; o += (size_t)GG * HH;   // mol agg
  float* GV1 = ws + o; o += GG;                // adg
  float* GV2 = ws + o; o += GG;                // mol segment sum

  const int NEG_INF_OI = (int)0x807FFFFF;      // f2oi(-inf)
  const dim3 B256(256);
  const unsigned gNg   = cdiv(NN, 512);        // gemm: 8 waves x 4 tiles x 16 rows
  const unsigned gGgm  = cdiv(GG, 512);
  const unsigned gN16  = cdiv(NN, 128);        // gru: 8 waves x 16 rows
  const unsigned gG16  = cdiv(GG, 128);
  const unsigned gE    = cdiv(EE, 256);
  const unsigned gNn   = cdiv(NN, 256);
  const unsigned gGg   = cdiv(GG, 256);
  const unsigned gNf   = cdiv((long long)NN * HH, 256);
  const unsigned gGf   = cdiv((long long)GG * HH, 256);
  const unsigned gEs   = cdiv((long long)EE * 32, 256);
  const unsigned gNs   = cdiv((long long)NN * 32, 256);

  // ---- lin1: x0 = leaky(x @ lin1_W^T + b) ----
  gemm_act_kernel<64, 64, 1, true, 4><<<gNg, B256, 0, stream>>>(x, lin1_W, 64, lin1_b, NB0, NN);

  // ---- GATEConv ----
  gemm_act_kernel<64, 64, 0, false, 4><<<gNg, B256, 0, stream>>>(NB0, gate_W1, 80, nullptr, NB1, NN); // s1
  gemm_act_kernel<64, 64, 0, false, 4><<<gNg, B256, 0, stream>>>(NB0, gate_W2, 64, nullptr, NB2, NN); // xW2
  node_dot_kernel<<<gNn, B256, 0, stream>>>(NB0, gate_att_r, nullptr, V1, nullptr, NN);               // gr
  gate_edge_logits_kernel<<<gE, B256, 0, stream>>>(NB1, V1, edge_attr, src, dst, gate_W1, gate_att_l, E1, EE);
  fill_i32_kernel<<<gNn, B256, 0, stream>>>(VI, NEG_INF_OI, NN);
  seg_max_kernel<<<gE, B256, 0, stream>>>(E1, dst, VI, EE);
  fill_f32_kernel<<<gNn, B256, 0, stream>>>(SUM, 0.f, NN);
  seg_expsum_kernel<<<gE, B256, 0, stream>>>(E1, dst, VI, SUM, E2, EE);
  fill_f32_kernel<<<gNf, B256, 0, stream>>>(NB3, 0.f, NN * HH);
  gat_scatter_kernel<<<gEs, B256, 0, stream>>>(NB2, E2, SUM, src, dst, NB3, EE);
  elu_bias_kernel<<<gNf, B256, 0, stream>>>(NB3, gate_b, NN);
  gru_fused_kernel<true><<<gN16, B256, 0, stream>>>(NB3, NB0, gru0_Wih, gru0_Whh, gru0_bih, gru0_bhh, NB1, NN); // x1

  // ---- atom GATConv ----
  gemm_act_kernel<64, 64, 0, false, 4><<<gNg, B256, 0, stream>>>(NB1, atom_W, 64, nullptr, NB2, NN);  // xs
  node_dot_kernel<<<gNn, B256, 0, stream>>>(NB2, atom_att_src, atom_att_dst, V1, V2, NN);
  edge_ss_logits_kernel<<<gE, B256, 0, stream>>>(V1, V2, src, dst, E1, EE);
  fill_i32_kernel<<<gNn, B256, 0, stream>>>(VI, NEG_INF_OI, NN);
  seg_max_kernel<<<gE, B256, 0, stream>>>(E1, dst, VI, EE);
  fill_f32_kernel<<<gNn, B256, 0, stream>>>(SUM, 0.f, NN);
  seg_expsum_kernel<<<gE, B256, 0, stream>>>(E1, dst, VI, SUM, E2, EE);
  fill_f32_kernel<<<gNf, B256, 0, stream>>>(NB3, 0.f, NN * HH);
  gat_scatter_kernel<<<gEs, B256, 0, stream>>>(NB2, E2, SUM, src, dst, NB3, EE);
  elu_bias_kernel<<<gNf, B256, 0, stream>>>(NB3, atom_b, NN);
  gru_fused_kernel<true><<<gN16, B256, 0, stream>>>(NB3, NB1, gru1_Wih, gru1_Whh, gru1_bih, gru1_bhh, NB0, NN); // x2

  // ---- attentive readout ----
  gemm_act_kernel<64, 64, 0, false, 4><<<gNg, B256, 0, stream>>>(NB0, mol_W, 64, nullptr, NB2, NN);   // xs (fixed per t)
  node_dot_kernel<<<gNn, B256, 0, stream>>>(NB2, mol_att_src, nullptr, V1, nullptr, NN);              // asn
  fill_f32_kernel<<<gGf, B256, 0, stream>>>(OUT, 0.f, GG * HH);
  pool_kernel<<<gNs, B256, 0, stream>>>(NB0, batch, OUT, NN);
  relu_kernel<<<gGf, B256, 0, stream>>>(OUT, GG * HH);

  for (int t = 0; t < 2; t++) {
    gemm_act_kernel<64, 64, 0, false, 4><<<gGgm, B256, 0, stream>>>(OUT, mol_W, 64, nullptr, GD, GG); // xd
    node_dot_kernel<<<gGg, B256, 0, stream>>>(GD, mol_att_dst, nullptr, GV1, nullptr, GG);            // adg
    mol_node_logits_kernel<<<gNn, B256, 0, stream>>>(V1, GV1, batch, V2, NN);
    fill_i32_kernel<<<gGg, B256, 0, stream>>>(VI, NEG_INF_OI, GG);
    seg_max_kernel<<<gNn, B256, 0, stream>>>(V2, batch, VI, NN);
    fill_f32_kernel<<<gGg, B256, 0, stream>>>(GV2, 0.f, GG);
    seg_expsum_kernel<<<gNn, B256, 0, stream>>>(V2, batch, VI, GV2, V3, NN);
    fill_f32_kernel<<<gGf, B256, 0, stream>>>(GA, 0.f, GG * HH);
    gat_scatter_kernel<<<gNs, B256, 0, stream>>>(NB2, V3, GV2, nullptr, batch, GA, NN);
    elu_bias_kernel<<<gGf, B256, 0, stream>>>(GA, mol_b, GG);
    gru_fused_kernel<true><<<gG16, B256, 0, stream>>>(GA, OUT, molg_Wih, molg_Whh, molg_bih, molg_bhh, OUT, GG);
  }

  lin2_kernel<<<gGg, B256, 0, stream>>>(OUT, lin2_W, lin2_b, (float*)d_out, GG);
}
